// PathGNNEncoder_37984690766250
// MI455X (gfx1250) — compile-verified, matrix operand order checked by reading, not run on hardware
//
#include <hip/hip_runtime.h>
#include <hip/hip_bf16.h>

typedef _Float16 f16;
typedef __attribute__((ext_vector_type(4)))  _Float16 v4h;
typedef __attribute__((ext_vector_type(16))) _Float16 v16h;
typedef __attribute__((ext_vector_type(8)))  float    v8f;

#define DIM 64

// ---------------------------------------------------------------------------
// WMMA fragment helpers (wave32, 16x16x32 f16 -> f32)
// Both A and (pre-transposed) B tiles are stored so each lane's fragment is
// two contiguous 16-byte groups -> two ds_load_b128 per fragment.
//   lane < 16 : M/N = lane,      K in {k0..k0+7, k0+16..k0+23}
//   lane >= 16: M/N = lane - 16, K in {k0+8..k0+15, k0+24..k0+31}
// C: v8f, element r maps to (M = r or r+8, N = lane&15).  (ISA 7.12.2)
// ---------------------------------------------------------------------------
__device__ inline v16h load_frag_k(const f16* rowbase, int k0, int lane) {
    const int sel = (lane >> 4) & 1;
    union { uint4 u[2]; v16h h; } cv;
    cv.u[0] = *(const uint4*)(rowbase + k0 + sel * 8);
    cv.u[1] = *(const uint4*)(rowbase + k0 + 16 + sel * 8);
    return cv.h;
}

__device__ inline v8f wmma_f16(v16h a, v16h b, v8f c) {
    return __builtin_amdgcn_wmma_f32_16x16x32_f16(false, a, false, b, (short)0, c,
                                                  false, false);
}

__device__ inline void store_v4h(f16* dst, float4 v) {
    v4h t;
    t[0] = (f16)v.x; t[1] = (f16)v.y; t[2] = (f16)v.z; t[3] = (f16)v.w;
    *(v4h*)dst = t;
}

// ---------------------------------------------------------------------------
// Utility kernels
// ---------------------------------------------------------------------------
__global__ void zero_f32(float* p, long long n) {
    long long t = (long long)blockIdx.x * blockDim.x + threadIdx.x;
    if (t < n) p[t] = 0.0f;
}

__global__ void deg_kernel(const long long* dst, float* deg, int E) {
    int e = blockIdx.x * blockDim.x + threadIdx.x;
    if (e < E) atomicAdd(&deg[(int)dst[e]], 1.0f);
}

__global__ void inv_deg_kernel(float* deg, int n) {
    int i = blockIdx.x * blockDim.x + threadIdx.x;
    if (i < n) deg[i] = 1.0f / fmaxf(deg[i], 1.0f);
}

// agg[dst] += x[src], one thread per (edge, dim) element -> coalesced
__global__ void scatter_add_kernel(const float* __restrict__ x,
                                   const long long* __restrict__ src,
                                   const long long* __restrict__ dst,
                                   float* __restrict__ agg, int E) {
    long long t = (long long)blockIdx.x * blockDim.x + threadIdx.x;
    if (t >= (long long)E * DIM) return;
    int e = (int)(t >> 6);
    int d = (int)(t & 63);
    atomicAdd(&agg[(size_t)dst[e] * DIM + d], x[(size_t)src[e] * DIM + d]);
}

// edge_emb[e] = (h[src]+h[dst]) * 0.5 , float4-vectorized
__global__ void edge_emb_kernel(const float* __restrict__ h,
                                const long long* __restrict__ src,
                                const long long* __restrict__ dst,
                                float* __restrict__ ee, int E) {
    long long t = (long long)blockIdx.x * blockDim.x + threadIdx.x;
    if (t >= (long long)E * (DIM / 4)) return;
    int e = (int)(t >> 4);
    int q = (int)(t & 15);
    const float4* h4 = (const float4*)h;
    float4 a = h4[(size_t)src[e] * 16 + q];
    float4 b = h4[(size_t)dst[e] * 16 + q];
    float4 r;
    r.x = (a.x + b.x) * 0.5f; r.y = (a.y + b.y) * 0.5f;
    r.z = (a.z + b.z) * 0.5f; r.w = (a.w + b.w) * 0.5f;
    ((float4*)ee)[(size_t)e * 16 + q] = r;
}

// ---------------------------------------------------------------------------
// Fused SAGE layer: out = relu( (agg*inv_deg) @ Wl + bl + x @ Wr )
// block = 128 threads (4 waves), one 16-node tile per block; each wave owns a
// 16-column tile. Weights staged TRANSPOSED in LDS (pitch 72 halves = 144 B,
// 16-B aligned) so B fragments are contiguous-along-k.
// ---------------------------------------------------------------------------
#define WP 72   // transposed weight pitch in halves

__global__ void sage_linear_kernel(const float* __restrict__ x,
                                   const float* __restrict__ agg,
                                   const float* __restrict__ inv_deg,
                                   const float* __restrict__ Wl,
                                   const float* __restrict__ Wr,
                                   const float* __restrict__ bl,
                                   float* __restrict__ out) {
    __shared__ f16 sX[16 * DIM];
    __shared__ f16 sG[16 * DIM];
    __shared__ f16 sWlT[DIM * WP];
    __shared__ f16 sWrT[DIM * WP];

    const int node0 = blockIdx.x * 16;
    const int tid   = threadIdx.x;
    const float4* Wl4 = (const float4*)Wl;
    const float4* Wr4 = (const float4*)Wr;

    // stage weights transposed: 1024 float4 each, 8 per thread
#pragma unroll
    for (int t = 0; t < 8; ++t) {
        int q = tid + 128 * t;            // vec4 index
        int row = q >> 4, colv = (q & 15) * 4;
        float4 a = Wl4[q];
        float4 b = Wr4[q];
        sWlT[(colv + 0) * WP + row] = (f16)a.x;
        sWlT[(colv + 1) * WP + row] = (f16)a.y;
        sWlT[(colv + 2) * WP + row] = (f16)a.z;
        sWlT[(colv + 3) * WP + row] = (f16)a.w;
        sWrT[(colv + 0) * WP + row] = (f16)b.x;
        sWrT[(colv + 1) * WP + row] = (f16)b.y;
        sWrT[(colv + 2) * WP + row] = (f16)b.z;
        sWrT[(colv + 3) * WP + row] = (f16)b.w;
    }
    // stage node tiles: 256 float4, 2 per thread; scale agg by inv_deg
#pragma unroll
    for (int t = 0; t < 2; ++t) {
        int q = tid + 128 * t;            // vec4 index
        int row = q >> 4, colv = (q & 15) * 4;
        float s = inv_deg[node0 + row];
        float4 xv = ((const float4*)x)[(size_t)(node0 + row) * 16 + (q & 15)];
        float4 gv = ((const float4*)agg)[(size_t)(node0 + row) * 16 + (q & 15)];
        gv.x *= s; gv.y *= s; gv.z *= s; gv.w *= s;
        store_v4h(&sX[row * DIM + colv], xv);
        store_v4h(&sG[row * DIM + colv], gv);
    }
    __syncthreads();

    const int w = tid >> 5, lane = tid & 31, n0 = w * 16;
    const f16* aRowX = sX + (lane & 15) * DIM;
    const f16* aRowG = sG + (lane & 15) * DIM;
    const f16* bRowL = sWlT + (size_t)(n0 + (lane & 15)) * WP;
    const f16* bRowR = sWrT + (size_t)(n0 + (lane & 15)) * WP;

    v8f c = {};
#pragma unroll
    for (int kk = 0; kk < DIM; kk += 32)
        c = wmma_f16(load_frag_k(aRowG, kk, lane), load_frag_k(bRowL, kk, lane), c);
#pragma unroll
    for (int kk = 0; kk < DIM; kk += 32)
        c = wmma_f16(load_frag_k(aRowX, kk, lane), load_frag_k(bRowR, kk, lane), c);

    const int n = n0 + (lane & 15);
    const float bias = bl[n];
#pragma unroll
    for (int r = 0; r < 8; ++r) {
        int m = (lane < 16) ? r : (r + 8);
        out[(size_t)(node0 + m) * DIM + n] = fmaxf(c[r] + bias, 0.0f);
    }
}

// ---------------------------------------------------------------------------
// Split-K path GEMM: psum += path_masks[p0:p0+32, kbeg:kend] @ edge_emb[kbeg:kend, :]
// block = 256 threads (8 waves): waves (w>>2) pick the 16-path sub-tile,
// (w&3) pick the 16-column tile. A row-major in LDS; edge_emb staged
// TRANSPOSED (pitch 136 halves = 272 B) so both fragments load as b128 pairs.
// ---------------------------------------------------------------------------
#define PT 32
#define KC 128
#define BP 136  // transposed B pitch in halves

__global__ void path_gemm_kernel(const float* __restrict__ pm,
                                 const float* __restrict__ ee,
                                 float* __restrict__ psum,
                                 int E, int klen) {
    __shared__ f16 sA[PT * KC];     // 8 KB : mask tile, row-major (paths x edges)
    __shared__ f16 sBt[DIM * BP];   // 17 KB: edge_emb tile, transposed (dims x edges)

    const int tid  = threadIdx.x;
    const int p0   = blockIdx.x * PT;
    const int kbeg = blockIdx.y * klen;
    const int kend = min(E, kbeg + klen);

    const int w = tid >> 5, lane = tid & 31;
    const int m0 = (w >> 2) * 16;        // path sub-tile
    const int n0 = (w & 3) * 16;         // column tile
    const f16* aRow = sA + (size_t)(m0 + (lane & 15)) * KC;
    const f16* bRow = sBt + (size_t)(n0 + (lane & 15)) * BP;
    v8f c = {};

    for (int k = kbeg; k < kend; k += KC) {
        const bool full = (k + KC <= kend);
        // stage A: 32x128 = 1024 float4, 4 per thread (b128 loads, b64 stores)
#pragma unroll
        for (int t = 0; t < 4; ++t) {
            int q = tid + 256 * t;                 // vec4 index
            int row = q >> 5, colv = (q & 31) * 4;
            int gk = k + colv;
            float4 v;
            if (full || gk + 3 < kend) {
                v = *(const float4*)&pm[(size_t)(p0 + row) * E + gk];
            } else {
                const float* pr = &pm[(size_t)(p0 + row) * E];
                v.x = (gk + 0 < kend) ? pr[gk + 0] : 0.0f;
                v.y = (gk + 1 < kend) ? pr[gk + 1] : 0.0f;
                v.z = (gk + 2 < kend) ? pr[gk + 2] : 0.0f;
                v.w = (gk + 3 < kend) ? pr[gk + 3] : 0.0f;
            }
            store_v4h(&sA[row * KC + colv], v);
        }
        // stage B transposed: 128x64 = 2048 float4, 8 per thread
#pragma unroll
        for (int t = 0; t < 8; ++t) {
            int q = tid + 256 * t;                 // vec4 index
            int row = q >> 4, colv = (q & 15) * 4; // row = edge-in-chunk, col = dim
            float4 v;
            if (full || k + row < kend) {
                v = ((const float4*)ee)[(size_t)(k + row) * 16 + (q & 15)];
            } else {
                v.x = v.y = v.z = v.w = 0.0f;
            }
            sBt[(colv + 0) * BP + row] = (f16)v.x;
            sBt[(colv + 1) * BP + row] = (f16)v.y;
            sBt[(colv + 2) * BP + row] = (f16)v.z;
            sBt[(colv + 3) * BP + row] = (f16)v.w;
        }
        __syncthreads();
#pragma unroll
        for (int kk = 0; kk < KC; kk += 32)
            c = wmma_f16(load_frag_k(aRow, kk, lane), load_frag_k(bRow, kk, lane), c);
        __syncthreads();
    }

    const int n = n0 + (lane & 15);
#pragma unroll
    for (int r = 0; r < 8; ++r) {
        int m = (lane < 16) ? r : (r + 8);
        atomicAdd(&psum[(size_t)(p0 + m0 + m) * DIM + n], c[r]);
    }
}

// ---------------------------------------------------------------------------
// Readout: out = psum @ Wro + bro   (512x64 @ 64x64)
// ---------------------------------------------------------------------------
__global__ void readout_kernel(const float* __restrict__ psum,
                               const float* __restrict__ Wro,
                               const float* __restrict__ bro,
                               float* __restrict__ out) {
    __shared__ f16 sP[16 * DIM];
    __shared__ f16 sWt[DIM * WP];
    const int row0 = blockIdx.x * 16;
    const int tid  = threadIdx.x;   // 128 threads
    const float4* W4 = (const float4*)Wro;

#pragma unroll
    for (int t = 0; t < 8; ++t) {
        int q = tid + 128 * t;            // vec4 index
        int row = q >> 4, colv = (q & 15) * 4;
        float4 a = W4[q];
        sWt[(colv + 0) * WP + row] = (f16)a.x;
        sWt[(colv + 1) * WP + row] = (f16)a.y;
        sWt[(colv + 2) * WP + row] = (f16)a.z;
        sWt[(colv + 3) * WP + row] = (f16)a.w;
    }
#pragma unroll
    for (int t = 0; t < 2; ++t) {
        int q = tid + 128 * t;
        int row = q >> 4, colv = (q & 15) * 4;
        float4 v = ((const float4*)psum)[(size_t)(row0 + row) * 16 + (q & 15)];
        store_v4h(&sP[row * DIM + colv], v);
    }
    __syncthreads();

    const int w = tid >> 5, lane = tid & 31, n0 = w * 16;
    const f16* aRow = sP + (lane & 15) * DIM;
    const f16* bRow = sWt + (size_t)(n0 + (lane & 15)) * WP;

    v8f c = {};
#pragma unroll
    for (int kk = 0; kk < DIM; kk += 32)
        c = wmma_f16(load_frag_k(aRow, kk, lane), load_frag_k(bRow, kk, lane), c);

    const int n = n0 + (lane & 15);
    const float bias = bro[n];
#pragma unroll
    for (int r = 0; r < 8; ++r) {
        int m = (lane < 16) ? r : (r + 8);
        out[(size_t)(row0 + m) * DIM + n] = c[r] + bias;
    }
}

// ---------------------------------------------------------------------------
// Host launcher
// ---------------------------------------------------------------------------
extern "C" void kernel_launch(void* const* d_in, const int* in_sizes, int n_in,
                              void* d_out, int out_size, void* d_ws, size_t ws_size,
                              hipStream_t stream) {
    const float*      nf  = (const float*)d_in[0];
    const long long*  ei  = (const long long*)d_in[1];
    const float*      pm  = (const float*)d_in[2];
    const float*      Wl0 = (const float*)d_in[3];
    const float*      Wr0 = (const float*)d_in[4];
    const float*      bl0 = (const float*)d_in[5];
    const float*      Wl1 = (const float*)d_in[6];
    const float*      Wr1 = (const float*)d_in[7];
    const float*      bl1 = (const float*)d_in[8];
    const float*      Wro = (const float*)d_in[9];
    const float*      bro = (const float*)d_in[10];
    float*            out = (float*)d_out;

    const int N = in_sizes[0] / DIM;        // 50000
    const int E = in_sizes[1] / 2;          // 400000
    const int P = in_sizes[2] / E;          // 512

    const long long* src = ei;
    const long long* dst = ei + E;

    // workspace layout (floats)
    float* ws = (float*)d_ws;
    size_t off = 0;
    float* deg  = ws + off; off += ((size_t)N + 63) & ~(size_t)63;
    float* agg  = ws + off; off += (size_t)N * DIM;
    float* h0   = ws + off; off += (size_t)N * DIM;
    float* h1   = ws + off; off += (size_t)N * DIM;
    float* ee   = ws + off; off += (size_t)E * DIM;
    float* psum = ws + off; off += (size_t)P * DIM;
    (void)ws_size; (void)n_in; (void)out_size;

    const int ZB = 256;
    auto blocks = [](long long n, int b) { return (unsigned)((n + b - 1) / b); };

    // degree -> inv_deg
    zero_f32<<<blocks(N, ZB), ZB, 0, stream>>>(deg, N);
    deg_kernel<<<blocks(E, ZB), ZB, 0, stream>>>(dst, deg, E);
    inv_deg_kernel<<<blocks(N, ZB), ZB, 0, stream>>>(deg, N);

    const long long EW = (long long)E * DIM;
    const unsigned sage_grid = (unsigned)(N / 16);   // 50000 = 16*3125 exactly

    // layer 0
    zero_f32<<<blocks((long long)N * DIM, ZB), ZB, 0, stream>>>(agg, (long long)N * DIM);
    scatter_add_kernel<<<blocks(EW, ZB), ZB, 0, stream>>>(nf, src, dst, agg, E);
    sage_linear_kernel<<<sage_grid, 128, 0, stream>>>(nf, agg, deg, Wl0, Wr0, bl0, h0);

    // layer 1
    zero_f32<<<blocks((long long)N * DIM, ZB), ZB, 0, stream>>>(agg, (long long)N * DIM);
    scatter_add_kernel<<<blocks(EW, ZB), ZB, 0, stream>>>(h0, src, dst, agg, E);
    sage_linear_kernel<<<sage_grid, 128, 0, stream>>>(h0, agg, deg, Wl1, Wr1, bl1, h1);

    // edge embeddings
    edge_emb_kernel<<<blocks((long long)E * (DIM / 4), ZB), ZB, 0, stream>>>(h1, src, dst, ee, E);

    // split-K path GEMM into psum
    zero_f32<<<blocks((long long)P * DIM, ZB), ZB, 0, stream>>>(psum, (long long)P * DIM);
    const int KSLICES = 96;
    const int klen = (E + KSLICES - 1) / KSLICES;
    dim3 pg_grid(P / PT, KSLICES);
    path_gemm_kernel<<<pg_grid, 256, 0, stream>>>(pm, ee, psum, E, klen);

    // readout
    readout_kernel<<<P / 16, 128, 0, stream>>>(psum, Wro, bro, out);
}
